// fact_Attention_69947837382898
// MI455X (gfx1250) — compile-verified, matrix operand order checked by reading, not run on hardware
//
#include <hip/hip_runtime.h>

// ---------------------------------------------------------------------------
// fact attention: energy = E @ F^T ; weights = softmax(energy, axis=1) ;
// outputs = weights @ F.   E:[N,D] F:[M,D] fp32.  N=M=8192, D=1024.
//
// d_out layout: outputs (N*D floats) then weights (N*M floats).
// The weights region doubles as the energy scratch (normalized in place).
//
// Precision: fp32 operands are split a = a_hi + a_lo (bf16 each); products
// accumulate a_hi*b_hi + a_lo*b_hi + a_hi*b_lo in fp32 WMMA accumulators
// (~fp32 accuracy at 3x bf16-WMMA cost; still ~8x cheaper per MAC than
// V_WMMA_F32_16X16X4_F32).
//
// Fast path (needs ~96 MB workspace): E,F pre-split to bf16 hi/lo row-major,
// F also transposed+split to [D][M] so both GEMMs use the NT fragment loader
// (pure b128 loads, zero conversion VALU in the inner loops).
// Fallback path (small d_ws): round-1 convert-in-loop kernels.
// ---------------------------------------------------------------------------

#define NDIM 8192
#define MDIM 8192
#define DDIM 1024

typedef __attribute__((ext_vector_type(16))) __bf16 v16bf;
typedef __attribute__((ext_vector_type(8)))  __bf16 v8bf;
typedef __attribute__((ext_vector_type(4)))  __bf16 v4bf;
typedef __attribute__((ext_vector_type(8)))  float  v8f;
typedef __attribute__((ext_vector_type(4)))  float  v4f;

__device__ __forceinline__ v8f wmma_bf16(v16bf a, v16bf b, v8f c) {
  // (neg_a, A, neg_b, B, c_mod, C, reuse_a, reuse_b)
  return __builtin_amdgcn_wmma_f32_16x16x32_bf16(false, a, false, b,
                                                 (short)0, c, false, false);
}

__device__ __forceinline__ void cvt_split16(const float* x, v16bf& hi, v16bf& lo) {
#pragma unroll
  for (int i = 0; i < 16; ++i) {
    float f  = x[i];
    __bf16 h = (__bf16)f;
    hi[i]    = h;
    lo[i]    = (__bf16)(f - (float)h);
  }
}

// --- fragment loaders -------------------------------------------------------
// ISA 16-bit A layout: lanes 0-15 hold row (lane), k = {0..7, 16..23};
// lanes 16-31 hold row (lane-16), k = {8..15, 24..31}.  B is the mirror
// (lane = column), so NT-GEMM B fragments use the same loader on row-major F.

// bf16 fragment from a pre-split row-major bf16 matrix: 2x b128, no VALU.
__device__ __forceinline__ void load_frag_bf16(const __bf16* __restrict__ p, int ld,
                                               int row, int k0, v16bf& f) {
  const int lane = threadIdx.x & 31;
  const __bf16* q = p + (size_t)(row + (lane & 15)) * ld + k0 + ((lane >> 4) << 3);
  v8bf* h = (v8bf*)&f;
  h[0] = *(const v8bf*)(q);
  h[1] = *(const v8bf*)(q + 16);
}

// fp32 fragment, split on the fly (fallback path + W operand of out-GEMM).
__device__ __forceinline__ void load_frag_rm(const float* __restrict__ p, int ld,
                                             int row, int k0, v16bf& hi, v16bf& lo) {
  const int lane = threadIdx.x & 31;
  const float* q = p + (size_t)(row + (lane & 15)) * ld + k0 + ((lane >> 4) << 3);
  float x[16];
  *(v4f*)(x + 0)  = *(const v4f*)(q + 0);
  *(v4f*)(x + 4)  = *(const v4f*)(q + 4);
  *(v4f*)(x + 8)  = *(const v4f*)(q + 16);
  *(v4f*)(x + 12) = *(const v4f*)(q + 20);
  cvt_split16(x, hi, lo);
}

// fp32 column fragment (fallback NN B): 16x b32, coalesced 64B per half-wave.
__device__ __forceinline__ void load_frag_cm(const float* __restrict__ p, int ld,
                                             int k0, int col0, v16bf& hi, v16bf& lo) {
  const int lane = threadIdx.x & 31;
  const float* q = p + (size_t)(k0 + ((lane >> 4) << 3)) * ld + col0 + (lane & 15);
  float x[16];
#pragma unroll
  for (int i = 0; i < 8; ++i) {
    x[i]     = q[(size_t)i * ld];
    x[i + 8] = q[(size_t)(i + 16) * ld];
  }
  cvt_split16(x, hi, lo);
}

// ---------------------------------------------------------------------------
// Pre-pass A: elementwise fp32 -> (bf16 hi, bf16 lo), row-major preserved.
// ---------------------------------------------------------------------------
__global__ __launch_bounds__(256) void split_rm_kernel(
    const float* __restrict__ in, __bf16* __restrict__ hi,
    __bf16* __restrict__ lo, int n4) {
  int i = blockIdx.x * 256 + threadIdx.x;
  if (i >= n4) return;
  v4f v = *(const v4f*)(in + (size_t)i * 4);
  v4bf h, l;
#pragma unroll
  for (int j = 0; j < 4; ++j) {
    __bf16 hh = (__bf16)v[j];
    h[j] = hh;
    l[j] = (__bf16)(v[j] - (float)hh);
  }
  *(v4bf*)(hi + (size_t)i * 4) = h;
  *(v4bf*)(lo + (size_t)i * 4) = l;
}

// ---------------------------------------------------------------------------
// Pre-pass B: F[M,D] -> Ft_hi/Ft_lo [D,M] (bf16), 64x64 LDS transpose.
// grid = (M/64, D/64).
// ---------------------------------------------------------------------------
__global__ __launch_bounds__(256) void transpose_split_kernel(
    const float* __restrict__ F, __bf16* __restrict__ tHi,
    __bf16* __restrict__ tLo) {
  __shared__ float tile[64][65];            // [d][m], pad kills bank conflicts
  const int m0 = blockIdx.x * 64;
  const int d0 = blockIdx.y * 64;
  const int tx = threadIdx.x & 63;
  const int ty = threadIdx.x >> 6;          // 0..3
#pragma unroll
  for (int i = 0; i < 16; ++i) {
    int m = i * 4 + ty;
    tile[tx][m] = F[(size_t)(m0 + m) * DDIM + d0 + tx];   // coalesced rows
  }
  __syncthreads();
#pragma unroll
  for (int i = 0; i < 16; ++i) {
    int d = i * 4 + ty;
    float f  = tile[d][tx];
    __bf16 h = (__bf16)f;
    tHi[(size_t)(d0 + d) * MDIM + m0 + tx] = h;           // coalesced rows
    tLo[(size_t)(d0 + d) * MDIM + m0 + tx] = (__bf16)(f - (float)h);
  }
}

// ---------------------------------------------------------------------------
// Kernel 1 (fast): energy[n,m] = sum_d E[n,d]*F[m,d]  (NT, pre-split bf16)
// WG = 8 waves (4x2), wave tile 32x32 (2x2 wmma), WG tile 128x64.
// Inner loop: 16x b128 loads, 12 wmma, no conversion VALU.
// ---------------------------------------------------------------------------
__global__ __launch_bounds__(256) void energy_gemm_fast_kernel(
    const __bf16* __restrict__ Ehi, const __bf16* __restrict__ Elo,
    const __bf16* __restrict__ Fhi, const __bf16* __restrict__ Flo,
    float* __restrict__ energy) {
  const int wave = threadIdx.x >> 5;
  const int lane = threadIdx.x & 31;
  const int n0 = blockIdx.y * 128 + (wave >> 1) * 32;
  const int m0 = blockIdx.x * 64  + (wave & 1) * 32;

  v8f acc[2][2] = {};

  for (int k0 = 0; k0 < DDIM; k0 += 32) {
    v16bf ah[2], al[2], bh[2], bl[2];
#pragma unroll
    for (int i = 0; i < 2; ++i) {
      load_frag_bf16(Ehi, DDIM, n0 + 16 * i, k0, ah[i]);
      load_frag_bf16(Elo, DDIM, n0 + 16 * i, k0, al[i]);
      load_frag_bf16(Fhi, DDIM, m0 + 16 * i, k0, bh[i]);
      load_frag_bf16(Flo, DDIM, m0 + 16 * i, k0, bl[i]);
    }
    // Three passes of 4 independent wmmas each (minimize RAW hazard NOPs).
#pragma unroll
    for (int i = 0; i < 2; ++i)
#pragma unroll
      for (int j = 0; j < 2; ++j) acc[i][j] = wmma_bf16(ah[i], bh[j], acc[i][j]);
#pragma unroll
    for (int i = 0; i < 2; ++i)
#pragma unroll
      for (int j = 0; j < 2; ++j) acc[i][j] = wmma_bf16(al[i], bh[j], acc[i][j]);
#pragma unroll
    for (int i = 0; i < 2; ++i)
#pragma unroll
      for (int j = 0; j < 2; ++j) acc[i][j] = wmma_bf16(ah[i], bl[j], acc[i][j]);
  }

#pragma unroll
  for (int i = 0; i < 2; ++i)
#pragma unroll
    for (int j = 0; j < 2; ++j) {
      const int n = n0 + 16 * i + ((lane >> 4) << 3);
      const int m = m0 + 16 * j + (lane & 15);
#pragma unroll
      for (int v = 0; v < 8; ++v)
        energy[(size_t)(n + v) * MDIM + m] = acc[i][j][v];
    }
}

// Kernel 1 (fallback): convert-in-loop version.
__global__ __launch_bounds__(256) void energy_gemm_fb_kernel(
    const float* __restrict__ E, const float* __restrict__ F,
    float* __restrict__ energy) {
  const int wave = threadIdx.x >> 5;
  const int lane = threadIdx.x & 31;
  const int n0 = blockIdx.y * 128 + (wave >> 1) * 32;
  const int m0 = blockIdx.x * 64  + (wave & 1) * 32;

  v8f acc[2][2] = {};
  for (int k0 = 0; k0 < DDIM; k0 += 32) {
    v16bf ah[2], al[2], bh[2], bl[2];
#pragma unroll
    for (int i = 0; i < 2; ++i) {
      load_frag_rm(E, DDIM, n0 + 16 * i, k0, ah[i], al[i]);
      load_frag_rm(F, DDIM, m0 + 16 * i, k0, bh[i], bl[i]);
    }
#pragma unroll
    for (int i = 0; i < 2; ++i)
#pragma unroll
      for (int j = 0; j < 2; ++j) acc[i][j] = wmma_bf16(ah[i], bh[j], acc[i][j]);
#pragma unroll
    for (int i = 0; i < 2; ++i)
#pragma unroll
      for (int j = 0; j < 2; ++j) acc[i][j] = wmma_bf16(al[i], bh[j], acc[i][j]);
#pragma unroll
    for (int i = 0; i < 2; ++i)
#pragma unroll
      for (int j = 0; j < 2; ++j) acc[i][j] = wmma_bf16(ah[i], bl[j], acc[i][j]);
  }
#pragma unroll
  for (int i = 0; i < 2; ++i)
#pragma unroll
    for (int j = 0; j < 2; ++j) {
      const int n = n0 + 16 * i + ((lane >> 4) << 3);
      const int m = m0 + 16 * j + (lane & 15);
#pragma unroll
      for (int v = 0; v < 8; ++v)
        energy[(size_t)(n + v) * MDIM + m] = acc[i][j][v];
    }
}

// ---------------------------------------------------------------------------
// Kernel 2: per-row max and sum(exp(e - max)).  One WG per row.
// ---------------------------------------------------------------------------
__global__ __launch_bounds__(256) void softmax_stats_kernel(
    const float* __restrict__ energy, float* __restrict__ stats) {
  const int row  = blockIdx.x;
  const int wave = threadIdx.x >> 5;
  const int lane = threadIdx.x & 31;
  const float* e = energy + (size_t)row * MDIM;

  float vals[32];
  float mx = -3.402823466e+38f;
#pragma unroll
  for (int i = 0; i < 32; ++i) {
    vals[i] = e[threadIdx.x + i * 256];
    mx = fmaxf(mx, vals[i]);
  }
#pragma unroll
  for (int off = 16; off; off >>= 1) mx = fmaxf(mx, __shfl_xor(mx, off, 32));

  __shared__ float smax[8], ssum[8];
  if (lane == 0) smax[wave] = mx;
  __syncthreads();
  float m = smax[0];
#pragma unroll
  for (int w = 1; w < 8; ++w) m = fmaxf(m, smax[w]);

  float s = 0.0f;
#pragma unroll
  for (int i = 0; i < 32; ++i) s += __expf(vals[i] - m);
#pragma unroll
  for (int off = 16; off; off >>= 1) s += __shfl_xor(s, off, 32);
  if (lane == 0) ssum[wave] = s;
  __syncthreads();
  if (threadIdx.x == 0) {
    float t = 0.0f;
#pragma unroll
    for (int w = 0; w < 8; ++w) t += ssum[w];
    stats[2 * row + 0] = m;
    stats[2 * row + 1] = t;
  }
}

// ---------------------------------------------------------------------------
// Kernel 3: in-place  w = exp(e - max) / sum.  grid = (M/1024, N).
// ---------------------------------------------------------------------------
__global__ __launch_bounds__(256) void softmax_norm_kernel(
    float* __restrict__ energy, const float* __restrict__ stats) {
  const int row = blockIdx.y;
  const int c   = (blockIdx.x * 256 + threadIdx.x) * 4;
  const float m   = stats[2 * row + 0];
  const float inv = 1.0f / stats[2 * row + 1];
  float4* p = (float4*)(energy + (size_t)row * MDIM + c);
  float4 v  = *p;
  v.x = __expf(v.x - m) * inv;
  v.y = __expf(v.y - m) * inv;
  v.z = __expf(v.z - m) * inv;
  v.w = __expf(v.w - m) * inv;
  *p = v;
}

// ---------------------------------------------------------------------------
// Kernel 4 (fast): outputs[n,d] = sum_m W[n,m]*F[m,d].
// WG = 16 rows x full D (W streamed from HBM exactly once);
// B fragments from pre-split transposed Ft[D,M] -> contiguous b128 loads.
// ---------------------------------------------------------------------------
__global__ __launch_bounds__(256) void out_gemm_fast_kernel(
    const float* __restrict__ W, const __bf16* __restrict__ FtHi,
    const __bf16* __restrict__ FtLo, float* __restrict__ out) {
  const int wave = threadIdx.x >> 5;
  const int lane = threadIdx.x & 31;
  const int n0 = blockIdx.x * 16;
  const int d0 = wave * 128;

  v8f acc[8] = {};
  for (int k0 = 0; k0 < MDIM; k0 += 32) {
    v16bf ah, al;
    load_frag_rm(W, MDIM, n0, k0, ah, al);      // only in-loop conversion left
#pragma unroll
    for (int t = 0; t < 8; ++t) {
      v16bf bh, bl;
      load_frag_bf16(FtHi, MDIM, d0 + 16 * t, k0, bh);
      load_frag_bf16(FtLo, MDIM, d0 + 16 * t, k0, bl);
      acc[t] = wmma_bf16(ah, bh, acc[t]);
      acc[t] = wmma_bf16(al, bh, acc[t]);
      acc[t] = wmma_bf16(ah, bl, acc[t]);
    }
  }

  const int n = n0 + ((lane >> 4) << 3);
#pragma unroll
  for (int t = 0; t < 8; ++t) {
    const int d = d0 + 16 * t + (lane & 15);
#pragma unroll
    for (int v = 0; v < 8; ++v)
      out[(size_t)(n + v) * DDIM + d] = acc[t][v];
  }
}

// Kernel 4 (fallback): strided-column B with convert-in-loop.
__global__ __launch_bounds__(256) void out_gemm_fb_kernel(
    const float* __restrict__ W, const float* __restrict__ F,
    float* __restrict__ out) {
  const int wave = threadIdx.x >> 5;
  const int lane = threadIdx.x & 31;
  const int n0 = blockIdx.x * 16;
  const int d0 = wave * 128;

  v8f acc[8] = {};
  for (int k0 = 0; k0 < MDIM; k0 += 32) {
    v16bf ah, al;
    load_frag_rm(W, MDIM, n0, k0, ah, al);
#pragma unroll
    for (int t = 0; t < 8; ++t) {
      v16bf bh, bl;
      load_frag_cm(F, DDIM, k0, d0 + 16 * t, bh, bl);
      acc[t] = wmma_bf16(ah, bh, acc[t]);
      acc[t] = wmma_bf16(al, bh, acc[t]);
      acc[t] = wmma_bf16(ah, bl, acc[t]);
    }
  }
  const int n = n0 + ((lane >> 4) << 3);
#pragma unroll
  for (int t = 0; t < 8; ++t) {
    const int d = d0 + 16 * t + (lane & 15);
#pragma unroll
    for (int v = 0; v < 8; ++v)
      out[(size_t)(n + v) * DDIM + d] = acc[t][v];
  }
}

// ---------------------------------------------------------------------------
extern "C" void kernel_launch(void* const* d_in, const int* in_sizes, int n_in,
                              void* d_out, int out_size, void* d_ws, size_t ws_size,
                              hipStream_t stream) {
  const float* E = (const float*)d_in[0];        // [N, D]
  const float* F = (const float*)d_in[1];        // [M, D]
  float* out     = (float*)d_out;                // [N, D]
  float* weights = out + (size_t)NDIM * DDIM;    // [N, M] (energy scratch, in-place)

  // Workspace layout: stats | Ehi | Elo | Fhi | Flo | FtHi | FtLo
  const size_t ED = (size_t)NDIM * DDIM;         // 8M elements
  const size_t FD = (size_t)MDIM * DDIM;         // 8M elements
  char*  ws    = (char*)d_ws;
  float* stats = (float*)ws;
  size_t off   = (size_t)2 * NDIM * sizeof(float);          // 64 KB
  __bf16* Ehi  = (__bf16*)(ws + off); off += ED * sizeof(__bf16);
  __bf16* Elo  = (__bf16*)(ws + off); off += ED * sizeof(__bf16);
  __bf16* Fhi  = (__bf16*)(ws + off); off += FD * sizeof(__bf16);
  __bf16* Flo  = (__bf16*)(ws + off); off += FD * sizeof(__bf16);
  __bf16* FtHi = (__bf16*)(ws + off); off += FD * sizeof(__bf16);
  __bf16* FtLo = (__bf16*)(ws + off); off += FD * sizeof(__bf16);
  const bool fast = (ws_size >= off);            // ws_size fixed -> deterministic

  dim3 g1(MDIM / 64, NDIM / 128);
  if (fast) {
    split_rm_kernel<<<(int)(ED / 4 / 256), 256, 0, stream>>>(E, Ehi, Elo, (int)(ED / 4));
    split_rm_kernel<<<(int)(FD / 4 / 256), 256, 0, stream>>>(F, Fhi, Flo, (int)(FD / 4));
    transpose_split_kernel<<<dim3(MDIM / 64, DDIM / 64), 256, 0, stream>>>(F, FtHi, FtLo);
    energy_gemm_fast_kernel<<<g1, 256, 0, stream>>>(Ehi, Elo, Fhi, Flo, weights);
  } else {
    energy_gemm_fb_kernel<<<g1, 256, 0, stream>>>(E, F, weights);
  }

  softmax_stats_kernel<<<NDIM, 256, 0, stream>>>(weights, stats);

  dim3 g3(MDIM / 1024, NDIM);
  softmax_norm_kernel<<<g3, 256, 0, stream>>>(weights, stats);

  if (fast) {
    out_gemm_fast_kernel<<<NDIM / 16, 256, 0, stream>>>(weights, FtHi, FtLo, out);
  } else {
    out_gemm_fb_kernel<<<NDIM / 16, 256, 0, stream>>>(weights, F, out);
  }
}
